// MultiheadAttention_45715631899147
// MI455X (gfx1250) — compile-verified, matrix operand order checked by reading, not run on hardware
//
#include <hip/hip_runtime.h>
#include <hip/hip_bf16.h>

typedef __attribute__((ext_vector_type(16))) _Float16 v16h;
typedef __attribute__((ext_vector_type(8)))  _Float16 h8;
typedef __attribute__((ext_vector_type(4)))  _Float16 h4;
typedef __attribute__((ext_vector_type(8)))  float    v8f;
typedef __attribute__((ext_vector_type(4)))  float    f4;
typedef __attribute__((ext_vector_type(4)))  unsigned int v4u;
typedef __attribute__((ext_vector_type(8)))  int      v8i;
typedef __attribute__((ext_vector_type(4)))  int      v4i;

constexpr int T_ = 2048, B_ = 2, E_ = 1024, H_ = 16, D_ = 64;
constexpr int M_ = T_ * B_;      // 4096 rows of (t,b)
constexpr int N3 = 3 * E_;       // 3072 packed qkv features
constexpr float SCALE = 0.125f;  // 1/sqrt(D)

// ---------------------------------------------------------------- WMMA helper
static __device__ __forceinline__ v8f wmma_f16(v16h a, v16h b, v8f c) {
  // D = A(16x32 f16) x B(32x16 f16) + C(16x16 f32)
  return __builtin_amdgcn_wmma_f32_16x16x32_f16(false, a, false, b, (short)0, c,
                                                false, false);
}

// Load a 16x32 (rows x K) fragment from a K-contiguous row-major operand.
// A-matrix ISA layout: lane(0-15)=row r, elems 0-7 = K(h*8..h*8+7),
// elems 8-15 = K(16+h*8..), h = lane>>4. B fragments are loaded from the
// transposed (N-major, K-contiguous) operand with the same pattern.
static __device__ __forceinline__ v16h load_frag(const _Float16* base, int row,
                                                 int ld, int k0, int lane) {
  const int r = lane & 15, hf = lane >> 4;
  const _Float16* p = base + (size_t)(row + r) * (size_t)ld + k0 + hf * 8;
  h8 lo = *(const h8*)(p);
  h8 hi = *(const h8*)(p + 16);
  v16h f;
#pragma unroll
  for (int i = 0; i < 8; ++i) { f[i] = lo[i]; f[i + 8] = hi[i]; }
  return f;
}

// ------------------------------------------- TDM: DMA a row of f32 into LDS
// Builds a minimal 2D D# (1 x nelem tile of 4-byte elements) and issues
// tensor_load_to_lds; waits on TENSORcnt. Each wave issues its own copy
// (TDM ignores EXEC; identical writes are benign), so no barrier is needed.
static __device__ __forceinline__ void tdm_load_row_f32(const float* gsrc,
                                                        float* lds_dst,
                                                        int nelem) {
#if __has_builtin(__builtin_amdgcn_tensor_load_to_lds) && \
    __has_builtin(__builtin_amdgcn_s_wait_tensorcnt)
  const unsigned lds_off = (unsigned)(unsigned long long)(void*)lds_dst;
  const unsigned long long ga = (unsigned long long)(const void*)gsrc;
  v4u g0;
  g0[0] = 1u;                                    // count=1: valid user D#
  g0[1] = lds_off;                               // lds_addr (low 32 = LDS off)
  g0[2] = (unsigned)(ga & 0xffffffffu);          // global_addr[31:0]
  g0[3] = (unsigned)((ga >> 32) & 0x01ffffffu)   // global_addr[56:32]
        | 0x80000000u;                           // type=2 ("image")
  v8i g1;
  g1[0] = (int)(2u << 16);                       // data_size=4B, wg_mask=0
  g1[1] = (int)((unsigned)nelem << 16);          // tensor_dim0[15:0]
  g1[2] = (int)(1u << 16);                       // tensor_dim1 = 1
  g1[3] = (int)((unsigned)nelem << 16);          // tile_dim0 = nelem
  g1[4] = 1;                                     // tile_dim1 = 1
  g1[5] = nelem;                                 // tensor_dim0_stride
  g1[6] = 0;
  g1[7] = 0;
  v4i gz = {0, 0, 0, 0};
#if defined(__clang_major__) && (__clang_major__ >= 23)
  v8i gz8 = {0, 0, 0, 0, 0, 0, 0, 0};
  __builtin_amdgcn_tensor_load_to_lds(g0, g1, gz, gz, gz8, 0);
#else
  __builtin_amdgcn_tensor_load_to_lds(g0, g1, gz, gz, 0);
#endif
  __builtin_amdgcn_s_wait_tensorcnt(0);
#else
  for (int i = (int)threadIdx.x; i < nelem; i += (int)blockDim.x)
    lds_dst[i] = gsrc[i];
  __syncthreads();
#endif
}

// ---------------------------------------------------------------- fp32 -> fp16
__global__ void k_cvt(const float* __restrict__ s, _Float16* __restrict__ d, int n) {
  int i = (blockIdx.x * blockDim.x + threadIdx.x) * 4;
  if (i < n) {
    f4 v = *(const f4*)(s + i);
    h4 o;
#pragma unroll
    for (int j = 0; j < 4; ++j) o[j] = (_Float16)v[j];
    *(h4*)(d + i) = o;
  }
}

// ------------------------------------------------- packed in-projection GEMM
// C[4096,3072] = X[4096,1024] * W[3072,1024]^T + bias, scattered into
// Qh/Kh: [B,H,T,D] f16 and Vt: [B,H,D,T] f16 (transposed for PV fragments).
// 2-deep ping-pong pipeline: no inter-buffer register copies.
__global__ void __launch_bounds__(128)
k_inproj(const _Float16* __restrict__ Xq, const _Float16* __restrict__ Xk,
         const _Float16* __restrict__ Xv, const _Float16* __restrict__ W,
         const float* __restrict__ bias, _Float16* __restrict__ Qh,
         _Float16* __restrict__ Kh, _Float16* __restrict__ Vt) {
  __shared__ __align__(16) float biasTile[64];
  const int lane = threadIdx.x & 31, wave = threadIdx.x >> 5;
  const int m0 = blockIdx.x * 64 + wave * 16;
  const int n0 = blockIdx.y * 64;
  const int sec = n0 >> 10;  // 0:q 1:k 2:v (64-wide tiles never straddle)
  const _Float16* X = (sec == 0) ? Xq : (sec == 1) ? Xk : Xv;

  tdm_load_row_f32(bias + n0, biasTile, 64);  // async-tensor bias staging

  v8f acc[4] = {};
  v16h a0 = load_frag(X, m0, E_, 0, lane);
  v16h b0[4], b1[4];
  v16h a1;
#pragma unroll
  for (int j = 0; j < 4; ++j) b0[j] = load_frag(W, n0 + j * 16, E_, 0, lane);

  int k = 32;
#pragma unroll 1
  for (int it = 0; it < 15; ++it, k += 64) {
    a1 = load_frag(X, m0, E_, k, lane);
#pragma unroll
    for (int j = 0; j < 4; ++j) b1[j] = load_frag(W, n0 + j * 16, E_, k, lane);
    __builtin_prefetch(X + (size_t)(m0 + (lane & 15)) * E_ + k + 128, 0, 0);
#pragma unroll
    for (int j = 0; j < 4; ++j) acc[j] = wmma_f16(a0, b0[j], acc[j]);

    a0 = load_frag(X, m0, E_, k + 32, lane);
#pragma unroll
    for (int j = 0; j < 4; ++j) b0[j] = load_frag(W, n0 + j * 16, E_, k + 32, lane);
#pragma unroll
    for (int j = 0; j < 4; ++j) acc[j] = wmma_f16(a1, b1[j], acc[j]);
  }
  // k == 992 here: last fragment set, then drain both buffers
  a1 = load_frag(X, m0, E_, k, lane);
#pragma unroll
  for (int j = 0; j < 4; ++j) b1[j] = load_frag(W, n0 + j * 16, E_, k, lane);
#pragma unroll
  for (int j = 0; j < 4; ++j) acc[j] = wmma_f16(a0, b0[j], acc[j]);
#pragma unroll
  for (int j = 0; j < 4; ++j) acc[j] = wmma_f16(a1, b1[j], acc[j]);

  const int hh = lane >> 4, cl = lane & 15;
#pragma unroll
  for (int j = 0; j < 4; ++j) {
    const int col = n0 + j * 16 + cl;
    const int cc = col & (E_ - 1);
    const int head = cc >> 6, dd = cc & (D_ - 1);
    const float bv = biasTile[j * 16 + cl];
#pragma unroll
    for (int i = 0; i < 8; ++i) {
      const int m = m0 + i + 8 * hh;
      const int t = m >> 1, b = m & 1;  // X rows are (t*B + b)
      const _Float16 hv = (_Float16)(acc[j][i] + bv);
      if (sec == 0)
        Qh[((size_t)(b * H_ + head) * T_ + t) * D_ + dd] = hv;
      else if (sec == 1)
        Kh[((size_t)(b * H_ + head) * T_ + t) * D_ + dd] = hv;
      else
        Vt[((size_t)(b * H_ + head) * D_ + dd) * T_ + t] = hv;
    }
  }
}

// ------------------------------------------------- flash attention (per b,h)
// One wave owns 16 query rows; online softmax over S in 32-wide blocks.
__global__ void __launch_bounds__(256)
k_attn(const _Float16* __restrict__ Qh, const _Float16* __restrict__ Kh,
       const _Float16* __restrict__ Vt, _Float16* __restrict__ attn,
       float* __restrict__ mbuf, float* __restrict__ lbuf) {
  __shared__ __align__(16) _Float16 plds[8][16 * 32];  // 1KB per wave
  const int lane = threadIdx.x & 31, wave = threadIdx.x >> 5;
  const int b = blockIdx.z, h = blockIdx.y;
  const int t0 = (blockIdx.x * 8 + wave) * 16;
  const _Float16* Qb = Qh + (size_t)(b * H_ + h) * T_ * D_;
  const _Float16* Kb = Kh + (size_t)(b * H_ + h) * T_ * D_;
  const _Float16* Vb = Vt + (size_t)(b * H_ + h) * D_ * T_;
  _Float16* myp = plds[wave];
  const int hh = lane >> 4, cl = lane & 15;

  const v16h qa0 = load_frag(Qb, t0, D_, 0, lane);
  const v16h qa1 = load_frag(Qb, t0, D_, 32, lane);

  v8f o[4] = {};
  float mrow[8], lrow[8];
#pragma unroll
  for (int i = 0; i < 8; ++i) { mrow[i] = -3.0e38f; lrow[i] = 0.0f; }

  for (int s0 = 0; s0 < T_; s0 += 32) {
    // Issue all K and V fragment loads up front; V latency hides under softmax.
    v16h kb00 = load_frag(Kb, s0, D_, 0, lane);
    v16h kb01 = load_frag(Kb, s0, D_, 32, lane);
    v16h kb10 = load_frag(Kb, s0 + 16, D_, 0, lane);
    v16h kb11 = load_frag(Kb, s0 + 16, D_, 32, lane);
    v16h vb[4];
#pragma unroll
    for (int j = 0; j < 4; ++j) vb[j] = load_frag(Vb, j * 16, T_, s0, lane);
    __builtin_prefetch(Kb + (size_t)(s0 + 32 + (lane & 15)) * D_, 0, 0);

    v8f c0 = {}, c1 = {};
    c0 = wmma_f16(qa0, kb00, c0);
    c0 = wmma_f16(qa1, kb01, c0);
    c1 = wmma_f16(qa0, kb10, c1);
    c1 = wmma_f16(qa1, kb11, c1);

    float corr[8];
#pragma unroll
    for (int i = 0; i < 8; ++i) {
      float a0 = c0[i] * SCALE, a1 = c1[i] * SCALE;
      float mx = fmaxf(a0, a1);  // row lives in one 16-lane half
#pragma unroll
      for (int msk = 1; msk < 16; msk <<= 1) mx = fmaxf(mx, __shfl_xor(mx, msk, 32));
      const float nm = fmaxf(mrow[i], mx);
      corr[i] = __expf(mrow[i] - nm);
      mrow[i] = nm;
      a0 = __expf(a0 - nm);
      a1 = __expf(a1 - nm);
      c0[i] = a0; c1[i] = a1;
      float sm = a0 + a1;
#pragma unroll
      for (int msk = 1; msk < 16; msk <<= 1) sm += __shfl_xor(sm, msk, 32);
      lrow[i] = lrow[i] * corr[i] + sm;
    }
#pragma unroll
    for (int j = 0; j < 4; ++j)
#pragma unroll
      for (int i = 0; i < 8; ++i) o[j][i] *= corr[i];

    // C-layout -> A-layout transpose of P through wave-private LDS
#pragma unroll
    for (int i = 0; i < 8; ++i) {
      const int r = i + 8 * hh;
      myp[r * 32 + cl] = (_Float16)c0[i];
      myp[r * 32 + 16 + cl] = (_Float16)c1[i];
    }
    asm volatile("s_wait_dscnt 0x0" ::: "memory");
    const v16h pa = load_frag(myp, 0, 32, 0, lane);
#pragma unroll
    for (int j = 0; j < 4; ++j)  // O[16,64] += P[16,32] x V[32,64]
      o[j] = wmma_f16(pa, vb[j], o[j]);
  }
  // epilogue: normalize, store attn in [T,B,E] f16 for the out-proj GEMM
#pragma unroll
  for (int j = 0; j < 4; ++j)
#pragma unroll
    for (int i = 0; i < 8; ++i) {
      const int t = t0 + i + 8 * hh;
      const int col = h * D_ + j * 16 + cl;
      attn[((size_t)t * B_ + b) * E_ + col] = (_Float16)(o[j][i] / lrow[i]);
    }
  if (cl == 0) {
#pragma unroll
    for (int i = 0; i < 8; ++i) {
      const int t = t0 + i + 8 * hh;
      mbuf[(size_t)(b * H_ + h) * T_ + t] = mrow[i];
      lbuf[(size_t)(b * H_ + h) * T_ + t] = lrow[i];
    }
  }
}

// ------------------------------------- averaged attention weights (B,T,S)
__global__ void __launch_bounds__(128)
k_avgw(const _Float16* __restrict__ Qh, const _Float16* __restrict__ Kh,
       const float* __restrict__ mbuf, const float* __restrict__ lbuf,
       float* __restrict__ avg) {
  const int lane = threadIdx.x & 31, wave = threadIdx.x >> 5;
  const int b = blockIdx.y;
  const int t0 = (blockIdx.x * 4 + wave) * 16;
  const int hh = lane >> 4, cl = lane & 15;
  float* avgb = avg + (size_t)b * T_ * T_;
  for (int s0 = 0; s0 < T_; s0 += 32) {
    v8f acc0 = {}, acc1 = {};
    for (int h = 0; h < H_; ++h) {
      const _Float16* Qb = Qh + (size_t)(b * H_ + h) * T_ * D_;
      const _Float16* Kb = Kh + (size_t)(b * H_ + h) * T_ * D_;
      const v16h qa0 = load_frag(Qb, t0, D_, 0, lane);
      const v16h qa1 = load_frag(Qb, t0, D_, 32, lane);
      v8f c0 = {}, c1 = {};
      c0 = wmma_f16(qa0, load_frag(Kb, s0, D_, 0, lane), c0);
      c0 = wmma_f16(qa1, load_frag(Kb, s0, D_, 32, lane), c0);
      c1 = wmma_f16(qa0, load_frag(Kb, s0 + 16, D_, 0, lane), c1);
      c1 = wmma_f16(qa1, load_frag(Kb, s0 + 16, D_, 32, lane), c1);
#pragma unroll
      for (int i = 0; i < 8; ++i) {
        const int t = t0 + i + 8 * hh;
        const float mm = mbuf[(size_t)(b * H_ + h) * T_ + t];
        const float il = (1.0f / (float)H_) / lbuf[(size_t)(b * H_ + h) * T_ + t];
        acc0[i] += __expf(c0[i] * SCALE - mm) * il;
        acc1[i] += __expf(c1[i] * SCALE - mm) * il;
      }
    }
#pragma unroll
    for (int i = 0; i < 8; ++i) {
      const int t = t0 + i + 8 * hh;
      avgb[(size_t)t * T_ + s0 + cl] = acc0[i];
      avgb[(size_t)t * T_ + s0 + 16 + cl] = acc1[i];
    }
  }
}

// ------------------------------------------------- out-projection GEMM
__global__ void __launch_bounds__(128)
k_outproj(const _Float16* __restrict__ A, const _Float16* __restrict__ W,
          const float* __restrict__ bias, float* __restrict__ out) {
  __shared__ __align__(16) float biasTile[64];
  const int lane = threadIdx.x & 31, wave = threadIdx.x >> 5;
  const int m0 = blockIdx.x * 64 + wave * 16;
  const int n0 = blockIdx.y * 64;

  tdm_load_row_f32(bias + n0, biasTile, 64);  // async-tensor bias staging

  v8f acc[4] = {};
  v16h a0 = load_frag(A, m0, E_, 0, lane);
  v16h b0[4], b1[4];
  v16h a1;
#pragma unroll
  for (int j = 0; j < 4; ++j) b0[j] = load_frag(W, n0 + j * 16, E_, 0, lane);

  int k = 32;
#pragma unroll 1
  for (int it = 0; it < 15; ++it, k += 64) {
    a1 = load_frag(A, m0, E_, k, lane);
#pragma unroll
    for (int j = 0; j < 4; ++j) b1[j] = load_frag(W, n0 + j * 16, E_, k, lane);
    __builtin_prefetch(A + (size_t)(m0 + (lane & 15)) * E_ + k + 128, 0, 0);
#pragma unroll
    for (int j = 0; j < 4; ++j) acc[j] = wmma_f16(a0, b0[j], acc[j]);

    a0 = load_frag(A, m0, E_, k + 32, lane);
#pragma unroll
    for (int j = 0; j < 4; ++j) b0[j] = load_frag(W, n0 + j * 16, E_, k + 32, lane);
#pragma unroll
    for (int j = 0; j < 4; ++j) acc[j] = wmma_f16(a1, b1[j], acc[j]);
  }
  a1 = load_frag(A, m0, E_, k, lane);
#pragma unroll
  for (int j = 0; j < 4; ++j) b1[j] = load_frag(W, n0 + j * 16, E_, k, lane);
#pragma unroll
  for (int j = 0; j < 4; ++j) acc[j] = wmma_f16(a0, b0[j], acc[j]);
#pragma unroll
  for (int j = 0; j < 4; ++j) acc[j] = wmma_f16(a1, b1[j], acc[j]);

  const int hh = lane >> 4, cl = lane & 15;
#pragma unroll
  for (int j = 0; j < 4; ++j) {
    const int col = n0 + j * 16 + cl;
    const float bv = biasTile[j * 16 + cl];
#pragma unroll
    for (int i = 0; i < 8; ++i) {
      const int m = m0 + i + 8 * hh;
      out[(size_t)m * E_ + col] = acc[j][i] + bv;
    }
  }
}

// ----------------------------------------------------------------- launcher
extern "C" void kernel_launch(void* const* d_in, const int* in_sizes, int n_in,
                              void* d_out, int out_size, void* d_ws, size_t ws_size,
                              hipStream_t stream) {
  (void)in_sizes; (void)n_in; (void)out_size; (void)ws_size;
  const float* q  = (const float*)d_in[0];
  const float* k  = (const float*)d_in[1];
  const float* v  = (const float*)d_in[2];
  const float* wi = (const float*)d_in[3];
  const float* bi = (const float*)d_in[4];
  const float* wo = (const float*)d_in[5];
  const float* bo = (const float*)d_in[6];

  char* ws = (char*)d_ws;
  const size_t MB = 1024 * 1024;
  _Float16* Xq = (_Float16*)(ws + 0 * MB);    // 8 MB  [4096,1024] f16
  _Float16* Xk = (_Float16*)(ws + 8 * MB);    // 8 MB
  _Float16* Xv = (_Float16*)(ws + 16 * MB);   // 8 MB
  _Float16* Wi = (_Float16*)(ws + 24 * MB);   // 6 MB  [3072,1024] f16
  _Float16* Wo = (_Float16*)(ws + 30 * MB);   // 2 MB  [1024,1024] f16
  _Float16* Qh = (_Float16*)(ws + 32 * MB);   // 8 MB  [B,H,T,D]
  _Float16* Kh = (_Float16*)(ws + 40 * MB);   // 8 MB  [B,H,T,D]
  _Float16* Vt = (_Float16*)(ws + 48 * MB);   // 8 MB  [B,H,D,T]
  _Float16* At = (_Float16*)(ws + 56 * MB);   // 8 MB  [T,B,E] f16
  float* mbuf  = (float*)(ws + 64 * MB);                  // 256 KB [B,H,T]
  float* lbuf  = (float*)(ws + 64 * MB + 256 * 1024);     // 256 KB [B,H,T]

  float* out0 = (float*)d_out;                 // attn_output [T,B,E] f32
  float* avg  = out0 + (size_t)T_ * B_ * E_;   // avg_weights [B,T,S] f32

  const int nX = T_ * B_ * E_;      // 4194304
  const int nWi = 3 * E_ * E_;      // 3145728
  const int nWo = E_ * E_;          // 1048576
  k_cvt<<<nX / 1024, 256, 0, stream>>>(q, Xq, nX);
  k_cvt<<<nX / 1024, 256, 0, stream>>>(k, Xk, nX);
  k_cvt<<<nX / 1024, 256, 0, stream>>>(v, Xv, nX);
  k_cvt<<<nWi / 1024, 256, 0, stream>>>(wi, Wi, nWi);
  k_cvt<<<nWo / 1024, 256, 0, stream>>>(wo, Wo, nWo);

  k_inproj<<<dim3(M_ / 64, N3 / 64), 128, 0, stream>>>(Xq, Xk, Xv, Wi, bi, Qh, Kh, Vt);
  k_attn<<<dim3(T_ / 128, H_, B_), 256, 0, stream>>>(Qh, Kh, Vt, At, mbuf, lbuf);
  k_avgw<<<dim3(T_ / 64, B_), 128, 0, stream>>>(Qh, Kh, mbuf, lbuf, avg);
  k_outproj<<<dim3(M_ / 64, E_ / 64), 128, 0, stream>>>(At, Wo, bo, out0);
}
